// GCM_77464030151163
// MI455X (gfx1250) — compile-verified, matrix operand order checked by reading, not run on hardware
//
#include <hip/hip_runtime.h>
#include <hip/hip_bf16.h>

typedef __attribute__((ext_vector_type(16))) _Float16 v16h;
typedef __attribute__((ext_vector_type(8)))  _Float16 v8h;
typedef __attribute__((ext_vector_type(8)))  float    v8f;
typedef __attribute__((ext_vector_type(4)))  int      v4i;

#define HD   32
#define DIMF 64
#define NK   27
#define MAXB 16
#define SCW  (NK * 2 * 32 * 16)  // 27648 halves per swizzled sconv weight set

#if defined(__has_builtin)
#if __has_builtin(__builtin_amdgcn_global_load_async_to_lds_b128) && \
    __has_builtin(__builtin_amdgcn_s_wait_asynccnt)
#define HAVE_ASYNC_LDS 1
#endif
#endif
#ifndef HAVE_ASYNC_LDS
#define HAVE_ASYNC_LDS 0
#endif

// K index pattern for 16-bit A/B WMMA fragments (ISA 7.12.2):
// lane<16 holds K {0..7,16..23}; lane>=16 holds K {8..15,24..31}
__device__ __forceinline__ int kpat(int lane, int j) {
  return ((lane >> 4) << 3) + (j & 7) + ((j >> 3) << 4);
}

__device__ __forceinline__ v8f wmma16(v16h a, v16h b, v8f c) {
  return __builtin_amdgcn_wmma_f32_16x16x32_f16(
      false, a, false, b, (short)0, c, false, false);
}

__device__ __forceinline__ v16h concat8(v8h lo, v8h hi) {
  return __builtin_shufflevector(lo, hi, 0, 1, 2, 3, 4, 5, 6, 7,
                                 8, 9, 10, 11, 12, 13, 14, 15);
}

__device__ __forceinline__ v16h loadA_f16(const _Float16* __restrict__ rowp,
                                          int lane) {
  const _Float16* p = rowp + ((lane >> 4) << 3);
  return concat8(*(const v8h*)(p), *(const v8h*)(p + 16));
}

__device__ __forceinline__ v16h loadA_f32(const float* __restrict__ rowp,
                                          int lane) {
  v16h a;
  const float* p = rowp + ((lane >> 4) << 3);
#pragma unroll
  for (int j = 0; j < 8; ++j) a[j] = (_Float16)p[j];
#pragma unroll
  for (int j = 0; j < 8; ++j) a[8 + j] = (_Float16)p[16 + j];
  return a;
}

__device__ __forceinline__ v16h zeroA() {
  v16h a;
#pragma unroll
  for (int j = 0; j < 16; ++j) a[j] = (_Float16)0.0f;
  return a;
}

// Contiguous global -> LDS fill of pre-swizzled weights.
// Async DMA path (CDNA5 GLOBAL_LOAD_ASYNC_TO_LDS_B128, ASYNCcnt) if available.
__device__ __forceinline__ void lds_fill_issue(_Float16* dst,
                                               const _Float16* src, int n,
                                               int tid, int nthreads) {
#if HAVE_ASYNC_LDS
  typedef __attribute__((address_space(1))) v4i gv4i_t;
  typedef __attribute__((address_space(3))) v4i lv4i_t;
  for (int t = tid * 8; t < n; t += nthreads * 8)
    __builtin_amdgcn_global_load_async_to_lds_b128(
        (gv4i_t*)(src + t), (lv4i_t*)(dst + t), 0, 0);
#else
  for (int t = tid * 8; t < n; t += nthreads * 8)
    *(v8h*)(dst + t) = *(const v8h*)(src + t);
#endif
}

__device__ __forceinline__ void lds_fill_wait() {
#if HAVE_ASYNC_LDS
  __builtin_amdgcn_s_wait_asynccnt(0);
#endif
}

// ---------------- one-shot weight pre-swizzle into fragment order -----------
__global__ void k_prep(const float* __restrict__ Wr1,
                       const float* __restrict__ Wr2,
                       const float* __restrict__ Wq1,
                       const float* __restrict__ Wq2,
                       const float* __restrict__ Wg1,
                       const float* __restrict__ Wg2,
                       const float* __restrict__ Wq3,
                       _Float16* __restrict__ sw) {
  int idx = blockIdx.x * blockDim.x + threadIdx.x;
  if (idx < 4 * SCW) {  // four sconv weight sets: (k, h, lane, j)
    int s = idx / SCW, t = idx % SCW;
    const float* W = (s == 0) ? Wr1 : (s == 1) ? Wr2 : (s == 2) ? Wq1 : Wq2;
    int j = t & 15, L = (t >> 4) & 31, h = (t >> 9) & 1, k = t >> 10;
    sw[idx] = (_Float16)W[(k * HD + kpat(L, j)) * HD + (h << 4) + (L & 15)];
  } else if (idx < 4 * SCW + 4096) {  // Wg1: (kh, nq, lane, j)
    int t = idx - 4 * SCW;
    int j = t & 15, L = (t >> 4) & 31, nq = (t >> 9) & 3, kh = t >> 11;
    sw[idx] = (_Float16)Wg1[(kh * 32 + kpat(L, j)) * DIMF + nq * 16 + (L & 15)];
  } else if (idx < 4 * SCW + 8192) {  // Wg2
    int t = idx - 4 * SCW - 4096;
    int j = t & 15, L = (t >> 4) & 31, nq = (t >> 9) & 3, kh = t >> 11;
    sw[idx] = (_Float16)Wg2[(kh * 32 + kpat(L, j)) * DIMF + nq * 16 + (L & 15)];
  } else if (idx < 4 * SCW + 8192 + 1024) {  // Wq3: (h, lane, j)
    int t = idx - 4 * SCW - 8192;
    int j = t & 15, L = (t >> 4) & 31, h = t >> 9;
    sw[idx] = (_Float16)Wq3[kpat(L, j) * HD + h * 16 + (L & 15)];
  }
}

// ---------------- input GEMM: y = x @ Wg1 + bg1 ; split into convx / glox ----
__global__ __launch_bounds__(128) void k_gemm_in(
    const float* __restrict__ X, const _Float16* __restrict__ Wsw,
    const float* __restrict__ bg1, float* __restrict__ convx,
    float* __restrict__ glox, _Float16* __restrict__ convxh,
    _Float16* __restrict__ gloxh, int N) {
  __shared__ alignas(32) _Float16 ldsW[2 * 4 * 32 * 16];
  const int tid = threadIdx.x;
  lds_fill_issue(ldsW, Wsw, 2 * 4 * 32 * 16, tid, blockDim.x);
  lds_fill_wait();
  __syncthreads();
  const v16h* Bf = (const v16h*)ldsW;
  const int lane = tid & 31;
  const int t0 = (blockIdx.x * 4 + (tid >> 5)) * 16;
  int nA = t0 + (lane & 15); if (nA > N - 1) nA = N - 1;

  v8f z8 = {0.f, 0.f, 0.f, 0.f, 0.f, 0.f, 0.f, 0.f};
  v8f acc[4] = {z8, z8, z8, z8};
#pragma unroll
  for (int kh = 0; kh < 2; ++kh) {
    v16h a = loadA_f32(X + (size_t)nA * DIMF + kh * 32, lane);
#pragma unroll
    for (int q = 0; q < 4; ++q)
      acc[q] = wmma16(a, Bf[(kh * 4 + q) * 32 + lane], acc[q]);
  }
  const int o = lane & 15;
#pragma unroll
  for (int r = 0; r < 8; ++r) {
    int node = t0 + ((lane >> 4) << 3) + r;
    if (node >= N) continue;
    size_t nb = (size_t)node * HD;
    float c0 = acc[0][r] + bg1[o];
    float c1 = acc[1][r] + bg1[16 + o];
    float g0 = acc[2][r] + bg1[32 + o];
    float g1 = acc[3][r] + bg1[48 + o];
    convx[nb + o]      = c0;  convxh[nb + o]      = (_Float16)c0;
    convx[nb + 16 + o] = c1;  convxh[nb + 16 + o] = (_Float16)c1;
    glox [nb + o]      = g0;  gloxh [nb + o]      = (_Float16)g0;
    glox [nb + 16 + o] = g1;  gloxh [nb + 16 + o] = (_Float16)g1;
  }
}

// ---------------- sconv epilogue (per 16-node tile) --------------------------
template <int MODE>
__device__ __forceinline__ void sconv_epi(
    v8f acc0, v8f acc1, int t0, int lane, int N, const float* __restrict__ bias,
    _Float16* __restrict__ out, const float* __restrict__ resid,
    float* __restrict__ m1, float* lds_bsum, float* lds_cnt,
    const int* __restrict__ batch_id) {
  const int o = lane & 15;
  const float b0 = bias[o], b1 = bias[16 + o];
#pragma unroll
  for (int r = 0; r < 8; ++r) {
    int node_raw = t0 + ((lane >> 4) << 3) + r;
    bool ok = node_raw < N;
    int node = ok ? node_raw : N - 1;
    size_t nb = (size_t)node * HD;
    float v0 = fmaxf(acc0[r] + b0, 0.f);
    float v1 = fmaxf(acc1[r] + b1, 0.f);
    if (MODE == 1) {
      float s0 = v0 + 2.f * resid[nb + o];
      float s1 = v1 + 2.f * resid[nb + 16 + o];
      if (ok) { out[nb + o] = (_Float16)s0; out[nb + 16 + o] = (_Float16)s1; }
    } else {
      if (ok) { out[nb + o] = (_Float16)v0; out[nb + 16 + o] = (_Float16)v1; }
    }
    if (MODE == 2) {
      float s = v0 + v1;
      s += __shfl_xor(s, 1, 32);
      s += __shfl_xor(s, 2, 32);
      s += __shfl_xor(s, 4, 32);
      s += __shfl_xor(s, 8, 32);
      if (ok && o == 0) m1[node] = s * (1.0f / (float)HD);
    }
    if (MODE == 3) {
      if (ok) {
        int b = batch_id[node] & (MAXB - 1);
        atomicAdd(&lds_bsum[b * HD + o], v0);
        atomicAdd(&lds_bsum[b * HD + 16 + o], v1);
        if (o == 0) atomicAdd(&lds_cnt[b], 1.0f);
      }
    }
  }
}

// ---------------- sparse conv: 2 tiles (32 nodes) per wave -------------------
// B fragments read once per k feed 4 WMMAs (2 tiles x 2 col-halves).
template <int MODE>
__global__ __launch_bounds__(128) void k_sconv(
    const _Float16* __restrict__ F, const _Float16* __restrict__ Wsw,
    const float* __restrict__ bias, const int* __restrict__ nbr,
    _Float16* __restrict__ out, const float* __restrict__ resid,
    float* __restrict__ m1, float* __restrict__ bsum,
    float* __restrict__ bcnt, const int* __restrict__ batch_id, int N) {
  __shared__ alignas(32) _Float16 ldsW[SCW];
  __shared__ float lds_bsum[MAXB * HD];
  __shared__ float lds_cnt[MAXB];
  const int tid = threadIdx.x;
  lds_fill_issue(ldsW, Wsw, SCW, tid, blockDim.x);
  if (MODE == 3) {
    for (int t = tid; t < MAXB * HD; t += blockDim.x) lds_bsum[t] = 0.f;
    for (int t = tid; t < MAXB; t += blockDim.x) lds_cnt[t] = 0.f;
  }
  lds_fill_wait();
  __syncthreads();
  const v16h* Bf = (const v16h*)ldsW;
  const int lane = tid & 31;
  const int t0 = (blockIdx.x * 4 + (tid >> 5)) * 32;  // tile pair base
  int nA0 = t0 + (lane & 15);      if (nA0 > N - 1) nA0 = N - 1;
  int nA1 = t0 + 16 + (lane & 15); if (nA1 > N - 1) nA1 = N - 1;

  v8f z8 = {0.f, 0.f, 0.f, 0.f, 0.f, 0.f, 0.f, 0.f};
  v8f acc00 = z8, acc01 = z8, acc10 = z8, acc11 = z8;
  const int* np0 = nbr + nA0;
  const int* np1 = nbr + nA1;
  for (int k = 0; k < NK; ++k, np0 += N, np1 += N) {
    int r0 = *np0;
    int r1 = *np1;
    v16h a0 = (r0 >= 0) ? loadA_f16(F + (size_t)r0 * HD, lane) : zeroA();
    v16h a1 = (r1 >= 0) ? loadA_f16(F + (size_t)r1 * HD, lane) : zeroA();
    v16h b0 = Bf[(k * 2 + 0) * 32 + lane];
    v16h b1 = Bf[(k * 2 + 1) * 32 + lane];
    acc00 = wmma16(a0, b0, acc00);
    acc01 = wmma16(a0, b1, acc01);
    acc10 = wmma16(a1, b0, acc10);
    acc11 = wmma16(a1, b1, acc11);
  }
  sconv_epi<MODE>(acc00, acc01, t0, lane, N, bias, out, resid, m1,
                  lds_bsum, lds_cnt, batch_id);
  sconv_epi<MODE>(acc10, acc11, t0 + 16, lane, N, bias, out, resid, m1,
                  lds_bsum, lds_cnt, batch_id);
  if (MODE == 3) {
    __syncthreads();
    for (int t = tid; t < MAXB * HD; t += blockDim.x)
      if (lds_bsum[t] != 0.f) atomicAdd(&bsum[t], lds_bsum[t]);
    for (int t = tid; t < MAXB; t += blockDim.x)
      if (lds_cnt[t] != 0.f) atomicAdd(&bcnt[t], lds_cnt[t]);
  }
}

// ---------------- batch mean finalize ---------------------------------------
__global__ void k_m2(const float* __restrict__ bsum,
                     const float* __restrict__ bcnt, float* __restrict__ m2) {
  int t = threadIdx.x;
  if (t < MAXB * HD) {
    float c = bcnt[t / HD];
    m2[t] = (c > 0.f) ? bsum[t] / c : 0.f;
  }
}

// ---------------- fused tail: enc + Wq3 GEMM + glo + Wg2 GEMM + residual ----
__global__ __launch_bounds__(128) void k_final(
    const float* __restrict__ X, const _Float16* __restrict__ convfh,
    const float* __restrict__ glox, const _Float16* __restrict__ out1h,
    const _Float16* __restrict__ out2h, const float* __restrict__ m1,
    const float* __restrict__ m2, const int* __restrict__ batch_id,
    const _Float16* __restrict__ swQ3, const float* __restrict__ bq3,
    const _Float16* __restrict__ swG2, const float* __restrict__ bg2,
    float* __restrict__ out, int N) {
  __shared__ alignas(32) _Float16 ldsQ3[2 * 32 * 16];
  __shared__ alignas(32) _Float16 ldsG2[2 * 4 * 32 * 16];
  __shared__ alignas(32) _Float16 stage[4][16 * HD];
  __shared__ float m2s[MAXB * HD];
  const int tid = threadIdx.x;
  lds_fill_issue(ldsQ3, swQ3, 2 * 32 * 16, tid, blockDim.x);
  lds_fill_issue(ldsG2, swG2, 2 * 4 * 32 * 16, tid, blockDim.x);
  for (int t = tid; t < MAXB * HD; t += blockDim.x) m2s[t] = m2[t];
  lds_fill_wait();
  __syncthreads();
  const v16h* BQ3 = (const v16h*)ldsQ3;
  const v16h* BG2 = (const v16h*)ldsG2;
  const int lane = tid & 31;
  const int wave = tid >> 5;
  const int t0 = (blockIdx.x * 4 + wave) * 16;
  int nA = t0 + (lane & 15); if (nA > N - 1) nA = N - 1;

  float m1v = m1[nA];
  int bA = batch_id[nA] & (MAXB - 1);
  const int cb = ((lane >> 4) << 3);
  v8h o1lo = *(const v8h*)(out1h + (size_t)nA * HD + cb);
  v8h o1hi = *(const v8h*)(out1h + (size_t)nA * HD + cb + 16);
  v8h o2lo = *(const v8h*)(out2h + (size_t)nA * HD + cb);
  v8h o2hi = *(const v8h*)(out2h + (size_t)nA * HD + cb + 16);
  v16h af;
#pragma unroll
  for (int j = 0; j < 16; ++j) {
    int c = kpat(lane, j);
    float o1 = (j < 8) ? (float)o1lo[j & 7] : (float)o1hi[j & 7];
    float o2 = (j < 8) ? (float)o2lo[j & 7] : (float)o2hi[j & 7];
    float e = sqrtf(m1v * m2s[bA * HD + c] + 1e-12f);
    af[j] = (_Float16)(e + o1 + o2);
  }
  v8f z8 = {0.f, 0.f, 0.f, 0.f, 0.f, 0.f, 0.f, 0.f};
  v8f f0 = wmma16(af, BQ3[lane], z8);
  v8f f1 = wmma16(af, BQ3[32 + lane], z8);

  const int o = lane & 15;
#pragma unroll
  for (int r = 0; r < 8; ++r) {
    int m = ((lane >> 4) << 3) + r;
    int node = t0 + m; if (node > N - 1) node = N - 1;
    size_t nb = (size_t)node * HD;
    float fv0 = fmaxf(f0[r] + bq3[o], 0.f);
    float fv1 = fmaxf(f1[r] + bq3[16 + o], 0.f);
    float g0 = fmaxf(glox[nb + o] - fv0, 0.f);
    float g1 = fmaxf(glox[nb + 16 + o] - fv1, 0.f);
    stage[wave][m * HD + o] = (_Float16)g0;
    stage[wave][m * HD + 16 + o] = (_Float16)g1;
  }
  __syncthreads();

  v16h a0 = loadA_f16(convfh + (size_t)nA * HD, lane);
  const _Float16* sp = &stage[wave][(lane & 15) * HD + cb];
  v16h a1 = concat8(*(const v8h*)sp, *(const v8h*)(sp + 16));

  v8f acc[4] = {z8, z8, z8, z8};
#pragma unroll
  for (int q = 0; q < 4; ++q) acc[q] = wmma16(a0, BG2[(0 * 4 + q) * 32 + lane], acc[q]);
#pragma unroll
  for (int q = 0; q < 4; ++q) acc[q] = wmma16(a1, BG2[(1 * 4 + q) * 32 + lane], acc[q]);

#pragma unroll
  for (int r = 0; r < 8; ++r) {
    int node = t0 + ((lane >> 4) << 3) + r;
    if (node >= N) continue;
    size_t nb = (size_t)node * DIMF;
#pragma unroll
    for (int q = 0; q < 4; ++q)
      out[nb + q * 16 + o] = X[nb + q * 16 + o] + acc[q][r] + bg2[q * 16 + o];
  }
}

extern "C" void kernel_launch(void* const* d_in, const int* in_sizes, int n_in,
                              void* d_out, int out_size, void* d_ws, size_t ws_size,
                              hipStream_t stream) {
  const float* x   = (const float*)d_in[0];
  const float* Wg1 = (const float*)d_in[1];
  const float* bg1 = (const float*)d_in[2];
  const float* Wg2 = (const float*)d_in[3];
  const float* bg2 = (const float*)d_in[4];
  const float* Wr1 = (const float*)d_in[5];
  const float* br1 = (const float*)d_in[6];
  const float* Wr2 = (const float*)d_in[7];
  const float* br2 = (const float*)d_in[8];
  const float* Wq1 = (const float*)d_in[9];
  const float* bq1 = (const float*)d_in[10];
  const float* Wq2 = (const float*)d_in[11];
  const float* bq2 = (const float*)d_in[12];
  const float* Wq3 = (const float*)d_in[13];
  const float* bq3 = (const float*)d_in[14];
  const int*   nbr = (const int*)d_in[15];
  const int*   bid = (const int*)d_in[16];
  const int N = in_sizes[0] / DIMF;

  char* ws = (char*)d_ws;
  const size_t featB = (size_t)N * HD * sizeof(float);
  const size_t halfB = (size_t)N * HD * sizeof(_Float16);
  float*    convx = (float*)(ws);
  float*    glox  = (float*)(ws + featB);
  _Float16* cvh   = (_Float16*)(ws + 2 * featB);             // reused as out2_h
  _Float16* glh   = (_Float16*)(ws + 2 * featB + halfB);
  _Float16* r1h   = (_Float16*)(ws + 2 * featB + 2 * halfB); // reused as out1_h
  _Float16* cfh   = (_Float16*)(ws + 2 * featB + 3 * halfB);
  float*    m1    = (float*)(ws + 2 * featB + 4 * halfB);
  float*    bsum  = m1 + N;
  float*    bcnt  = bsum + MAXB * HD;
  float*    m2    = bcnt + MAXB;
  _Float16* sw    = (_Float16*)(m2 + MAXB * HD);  // pre-swizzled weights
  _Float16* swR1  = sw;
  _Float16* swR2  = sw + SCW;
  _Float16* swQ1  = sw + 2 * SCW;
  _Float16* swQ2  = sw + 3 * SCW;
  _Float16* swG1  = sw + 4 * SCW;
  _Float16* swG2  = sw + 4 * SCW + 4096;
  _Float16* swQ3  = sw + 4 * SCW + 8192;
  _Float16* out1h = r1h;
  _Float16* out2h = cvh;

  (void)hipMemsetAsync(bsum, 0, (size_t)(MAXB * HD + MAXB) * sizeof(float),
                       stream);

  const int swTotal = 4 * SCW + 8192 + 1024;
  k_prep<<<(swTotal + 255) / 256, 256, 0, stream>>>(Wr1, Wr2, Wq1, Wq2,
                                                    Wg1, Wg2, Wq3, sw);

  const int tiles = (N + 15) / 16;
  const int blocks1 = (tiles + 3) / 4;            // 1 tile per wave kernels
  const int pairs = (tiles + 1) / 2;
  const int blocks2 = (pairs + 3) / 4;            // 2 tiles per wave (sconv)
  dim3 blk(128);

  k_gemm_in<<<blocks1, blk, 0, stream>>>(x, swG1, bg1, convx, glox, cvh, glh, N);
  k_sconv<0><<<blocks2, blk, 0, stream>>>(cvh, swR1, br1, nbr, r1h,
                                          nullptr, nullptr, nullptr, nullptr, nullptr, N);
  k_sconv<1><<<blocks2, blk, 0, stream>>>(r1h, swR2, br2, nbr, cfh,
                                          convx, nullptr, nullptr, nullptr, nullptr, N);
  k_sconv<2><<<blocks2, blk, 0, stream>>>(glh, swQ1, bq1, nbr, out1h,
                                          nullptr, m1, nullptr, nullptr, nullptr, N);
  k_sconv<3><<<blocks2, blk, 0, stream>>>(glh, swQ2, bq2, nbr, out2h,
                                          nullptr, nullptr, bsum, bcnt, bid, N);
  k_m2<<<1, MAXB * HD, 0, stream>>>(bsum, bcnt, m2);
  k_final<<<blocks1, blk, 0, stream>>>(x, cfh, glox, out1h, out2h, m1, m2, bid,
                                       swQ3, bq3, swG2, bg2, (float*)d_out, N);
}
